// HouseholderSequence_16578573763427
// MI455X (gfx1250) — matrix operand order repaired and checked by reading, not compiled, not run on hardware
//
#include <hip/hip_runtime.h>
#include <hip/hip_bf16.h>

// HouseholderSequence on MI455X (gfx1250):
//   out = A @ (H1 H2 ... H32),  Hk = I - (2/||qk||^2) qk qk^T
// Fold reflectors into M (256x256), then one bf16-split WMMA GEMM:
//   out ~= Ahi*Mhi + Ahi*Mlo + Alo*Mhi   (lo*lo term ~2^-16, dropped)
// Roofline: 34.4 GFLOP vs 512 MB HBM traffic (22 us floor @ 23.3 TB/s).
// v2: column-partitioned waves + 4x row-tile register blocking so each
// B-fragment register load feeds 12 WMMAs (was 3) -> B-pack L2 traffic
// drops from ~4 GB to ~1 GB and the loop is WMMA-issue bound.

#define N_TOT 262144
#define DDIM  256
#define KREF  32

typedef __attribute__((ext_vector_type(16))) __bf16         v16bf;
typedef __attribute__((ext_vector_type(8)))  float          v8f;
typedef __attribute__((ext_vector_type(8)))  unsigned short v8u16;
typedef __attribute__((ext_vector_type(4)))  unsigned short v4u16;

__device__ __forceinline__ unsigned short f2bf_rne(float f) {
  unsigned u = __float_as_uint(f);
  u += 0x7FFFu + ((u >> 16) & 1u);   // round-to-nearest-even to bf16
  return (unsigned short)(u >> 16);
}
__device__ __forceinline__ float bf2f(unsigned short h) {
  return __uint_as_float(((unsigned)h) << 16);
}

union FragAB { v8u16 u[2]; v16bf b; };  // 32B: 16 bf16 halves = one WMMA operand

// ---------------------------------------------------------------------------
// Kernel 0: build M = H1*H2*...*H32 row-by-row (rows are independent:
// M[i,:] <- M[i,:] - beta*(M[i,:].q) q^T), one row per wave32, 8 floats/lane.
// Emit bf16 hi/lo directly in this code's WMMA B-fragment order (K contiguous per lane half):
//   fragment(kc,nt): lane L<16 holds col nt*16+L, K = kc*32 + h (h=0..15)
//                    lane L>=16 holds col nt*16+L-16, K = kc*32 + 16 + h
// ---------------------------------------------------------------------------
__global__ __launch_bounds__(128)
void build_q_fragments(const float* __restrict__ qv,
                       unsigned short* __restrict__ bhi,
                       unsigned short* __restrict__ blo) {
  const int lane = threadIdx.x & 31;
  const int row  = blockIdx.x * 4 + (threadIdx.x >> 5);   // 0..255 = K index

  float r[8];
#pragma unroll
  for (int j = 0; j < 8; ++j) r[j] = ((lane + 32 * j) == row) ? 1.0f : 0.0f;

  for (int k = 0; k < KREF; ++k) {
    float q[8];
    float t = 0.0f, s = 0.0f;
#pragma unroll
    for (int j = 0; j < 8; ++j) {
      q[j] = qv[k * DDIM + lane + 32 * j];
      t += r[j] * q[j];
      s += q[j] * q[j];
    }
#pragma unroll
    for (int off = 16; off > 0; off >>= 1) {   // wave32 butterfly reduction
      t += __shfl_xor(t, off, 32);
      s += __shfl_xor(s, off, 32);
    }
    const float bt = (2.0f / s) * t;
#pragma unroll
    for (int j = 0; j < 8; ++j) r[j] -= bt * q[j];
  }

  const int kc     = row >> 5;
  const int local  = row & 31;
  const int h      = local & 15;
  const int laneHi = local >> 4;
#pragma unroll
  for (int j = 0; j < 8; ++j) {
    const int c  = lane + 32 * j;             // column of M
    const int nt = c >> 4;
    const int L  = laneHi * 16 + (c & 15);
    const unsigned short hi = f2bf_rne(r[j]);
    const unsigned short lo = f2bf_rne(r[j] - bf2f(hi));
    const size_t idx = ((size_t)((kc * 16 + nt) * 32 + L)) * 16 + h;
    bhi[idx] = hi;
    blo[idx] = lo;
  }
}

// ---------------------------------------------------------------------------
// Kernel 1: out[N,256] = A[N,256] @ M via v_wmma_f32_16x16x32_bf16.
// Block = 128 threads (4 waves), 64 rows of A staged in LDS as bf16 hi/lo.
// Waves partition the 16 column tiles (4 each); every wave register-blocks
// all 4 row tiles, so each B fragment in registers feeds 12 WMMAs.
// ---------------------------------------------------------------------------
__global__ __launch_bounds__(128)
void householder_gemm(const float* __restrict__ A,
                      const unsigned short* __restrict__ bhi,
                      const unsigned short* __restrict__ blo,
                      float* __restrict__ out) {
  __shared__ unsigned short sAhi[64 * DDIM];   // 32 KB
  __shared__ unsigned short sAlo[64 * DDIM];   // 32 KB

  const int  tid     = threadIdx.x;
  const int  blk     = blockIdx.x;
  const long rowBase = (long)blk * 64;

  // Stage + split: 64x256 fp32 -> bf16 hi/lo into LDS (coalesced float4).
  const float4* gA = (const float4*)(A + rowBase * DDIM);
  for (int it = tid; it < (64 * DDIM) / 4; it += 128) {
    __builtin_prefetch(gA + it + 256, 0, 1);   // global_prefetch_b8 (speculative)
    const float4 v = gA[it];
    const unsigned short h0 = f2bf_rne(v.x), h1 = f2bf_rne(v.y);
    const unsigned short h2 = f2bf_rne(v.z), h3 = f2bf_rne(v.w);
    const v4u16 vh = {h0, h1, h2, h3};
    const v4u16 vl = {f2bf_rne(v.x - bf2f(h0)), f2bf_rne(v.y - bf2f(h1)),
                      f2bf_rne(v.z - bf2f(h2)), f2bf_rne(v.w - bf2f(h3))};
    *(v4u16*)&sAhi[it * 4] = vh;
    *(v4u16*)&sAlo[it * 4] = vl;
  }
  if (tid < 64) out[(size_t)N_TOT * DDIM + rowBase + tid] = 0.0f;  // logabsdet
  __syncthreads();

  const int lane   = tid & 31;
  const int wave   = tid >> 5;          // 0..3 -> column tiles wave*4..wave*4+3
  const int mrow   = lane & 15;         // A-fragment M index / C column lane
  const int laneHi = lane >> 4;
  const int kb     = laneHi * 16;       // A layout: lanes>=16 start at K+16

  v8f acc[4][4];                        // [row tile][local column tile]
#pragma unroll
  for (int rt = 0; rt < 4; ++rt)
#pragma unroll
    for (int tl = 0; tl < 4; ++tl)
#pragma unroll
      for (int i = 0; i < 8; ++i) acc[rt][tl][i] = 0.0f;

  for (int kc = 0; kc < 8; ++kc) {
    const int k0 = kc * 32;
    // A fragments for all 4 row tiles (16x32 bf16, K contiguous per lane half, 2x ds_b128 each)
    FragAB ah[4], al[4];
#pragma unroll
    for (int rt = 0; rt < 4; ++rt) {
      const unsigned short* rp = &sAhi[(rt * 16 + mrow) * DDIM + k0 + kb];
      ah[rt].u[0] = *(const v8u16*)(rp);
      ah[rt].u[1] = *(const v8u16*)(rp + 8);
      const unsigned short* rq = &sAlo[(rt * 16 + mrow) * DDIM + k0 + kb];
      al[rt].u[0] = *(const v8u16*)(rq);
      al[rt].u[1] = *(const v8u16*)(rq + 8);
    }
#pragma unroll
    for (int tl = 0; tl < 4; ++tl) {
      const int nt = wave * 4 + tl;
      const size_t fo = ((size_t)((kc * 16 + nt) * 32 + lane)) * 16;
      FragAB bh, bl;
      bh.u[0] = *(const v8u16*)(bhi + fo);
      bh.u[1] = *(const v8u16*)(bhi + fo + 8);
      bl.u[0] = *(const v8u16*)(blo + fo);
      bl.u[1] = *(const v8u16*)(blo + fo + 8);
#pragma unroll
      for (int rt = 0; rt < 4; ++rt) {   // 12 WMMAs per B-fragment load
        acc[rt][tl] = __builtin_amdgcn_wmma_f32_16x16x32_bf16(
            false, ah[rt].b, false, bh.b, (short)0, acc[rt][tl], false, false);
        acc[rt][tl] = __builtin_amdgcn_wmma_f32_16x16x32_bf16(
            false, ah[rt].b, false, bl.b, (short)0, acc[rt][tl], false, false);
        acc[rt][tl] = __builtin_amdgcn_wmma_f32_16x16x32_bf16(
            false, al[rt].b, false, bh.b, (short)0, acc[rt][tl], false, false);
      }
    }
  }

  // Store C tiles (ISA C layout: VGPR r -> row +laneHi*8 + r, col fixed).
#pragma unroll
  for (int rt = 0; rt < 4; ++rt)
#pragma unroll
    for (int tl = 0; tl < 4; ++tl) {
      const int col = (wave * 4 + tl) * 16 + mrow;
      float* cp = out + (rowBase + rt * 16 + laneHi * 8) * DDIM + col;
#pragma unroll
      for (int rr = 0; rr < 8; ++rr) cp[(size_t)rr * DDIM] = acc[rt][tl][rr];
    }
}

extern "C" void kernel_launch(void* const* d_in, const int* in_sizes, int n_in,
                              void* d_out, int out_size, void* d_ws, size_t ws_size,
                              hipStream_t stream) {
  const float* A  = (const float*)d_in[0];   // [N, 256] fp32
  const float* Qv = (const float*)d_in[1];   // [32, 256] fp32
  float* out = (float*)d_out;                // [N*256] outputs ++ [N] logabsdet

  // Workspace: prepacked WMMA B fragments, bf16 hi (128 KB) + lo (128 KB).
  unsigned short* bhi = (unsigned short*)d_ws;
  unsigned short* blo = bhi + (size_t)8 * 16 * 32 * 16;

  build_q_fragments<<<64, 128, 0, stream>>>(Qv, bhi, blo);
  householder_gemm<<<N_TOT / 64, 128, 0, stream>>>(A, bhi, blo, out);
}
